// MHSA_43628277793485
// MI455X (gfx1250) — matrix operand order repaired and checked by reading, not compiled, not run on hardware
//
#include <hip/hip_runtime.h>

typedef __bf16 bf16;
typedef __attribute__((ext_vector_type(16))) __bf16 v16bf;
typedef __attribute__((ext_vector_type(8)))  __bf16 v8bf;
typedef __attribute__((ext_vector_type(8)))  float  v8f;

#define DIM 512
#define HEADS 16
#define HD 32
#define NTOK 1024
#define BATCH 8
#define M_TOTAL (BATCH * NTOK)            // 8192
// 32^-0.5 * log2(e): softmax done in exp2 domain
#define SCALE_Q 0.25503472525567663f
#define KCHUNK 256
#define APAD 40                            // padded row length for 32-wide tiles
#define VLD (KCHUNK + 8)                   // padded row length for V^T chunk

union V16 { v16bf v; v8bf h[2]; };

// CDNA5 async global->LDS copy helpers (ASYNCcnt path, CDNA5 ISA).
// IOFFSET applies to BOTH the LDS and the global address, so one address pair
// drives several b128 transfers.
#define ASYNC_LDS_B128_X1(lds, gaddr)                                          \
    asm volatile("global_load_async_to_lds_b128 %0, %1, off"                   \
                 :: "v"(lds), "v"(gaddr) : "memory")
#define ASYNC_LDS_B128_X2(lds, gaddr)                                          \
    asm volatile("global_load_async_to_lds_b128 %0, %1, off\n\t"               \
                 "global_load_async_to_lds_b128 %0, %1, off offset:16"         \
                 :: "v"(lds), "v"(gaddr) : "memory")
#define ASYNC_LDS_B128_X4(lds, gaddr)                                          \
    asm volatile("global_load_async_to_lds_b128 %0, %1, off\n\t"               \
                 "global_load_async_to_lds_b128 %0, %1, off offset:16\n\t"     \
                 "global_load_async_to_lds_b128 %0, %1, off offset:32\n\t"     \
                 "global_load_async_to_lds_b128 %0, %1, off offset:48"         \
                 :: "v"(lds), "v"(gaddr) : "memory")
#define WAIT_ASYNC0() asm volatile("s_wait_asynccnt 0" ::: "memory")

__device__ __forceinline__ unsigned lds_off(const void* p) {
    return (unsigned)(size_t)p;            // low 32 bits of generic addr = LDS offset
}

// ---------------- WMMA fragment loaders (documented gfx1250 layouts) ----------
// All loaders read K-contiguous memory -> 16-byte vector loads only.

// A-matrix 16x32 bf16, row-major, row stride ldm (elements).
// lane<16: row=lane, K={0..7,16..23}; lane>=16: row=lane-16, K={8..15,24..31}
__device__ __forceinline__ v16bf load_A(const bf16* base, int ldm) {
    const int lane = threadIdx.x & 31;
    const bf16* p = base + (lane & 15) * ldm + ((lane >> 4) * 8);
    V16 r;
    r.h[0] = *(const v8bf*)(p);        // K = koff .. koff+7
    r.h[1] = *(const v8bf*)(p + 16);   // K = koff+16 .. koff+23
    return r.v;
}

// B-matrix 32(K)x16(N) bf16 taken as M^T of row-major M[16 rows=N][32 cols=K],
// row stride ldr. lanes 0-15: K={0..7,16..23}, N=lane; lanes 16-31: K={8..15,24..31}, N=lane-16
__device__ __forceinline__ v16bf load_BT(const bf16* base, int ldr) {
    const int lane = threadIdx.x & 31;
    const bf16* p = base + (lane & 15) * ldr + ((lane >> 4) * 8);
    V16 r;
    r.h[0] = *(const v8bf*)(p);
    r.h[1] = *(const v8bf*)(p + 16);
    return r.v;
}

__device__ __forceinline__ v8f wmma_bf16(v16bf a, v16bf b, v8f c) {
    return __builtin_amdgcn_wmma_f32_16x16x32_bf16(false, a, false, b, (short)0, c,
                                                   false, false);
}

// ---------------- Kernel 1a: f32 -> bf16 convert (elementwise) ----------------

__global__ void k_cvt_bf16(const float* __restrict__ in, bf16* __restrict__ out, int n) {
    int i = blockIdx.x * blockDim.x + threadIdx.x;
    int stride = gridDim.x * blockDim.x;
    for (; i < n; i += stride) out[i] = (bf16)in[i];
}

// ---------------- Kernel 1b: f32 [512][N] -> bf16 transposed [N][512] ---------

__global__ void k_cvt_t(const float* __restrict__ in, bf16* __restrict__ out, int total) {
    int i = blockIdx.x * blockDim.x + threadIdx.x;
    int stride = gridDim.x * blockDim.x;
    const int N = total >> 9;             // in is [512][N]
    for (; i < total; i += stride) {
        int n = i >> 9;                   // out row (K=512 contiguous)
        int k = i & 511;
        out[i] = (bf16)in[(size_t)k * N + n];
    }
}

// ---------------- Kernel 2: QKV GEMM  [8192,512] x [512,1536] ----------------
// Block tile 64(M) x 128(N), 8 waves as 2x4, wave tile 32x32 (4 WMMA / k-step).
// Double-buffered LDS fed by async global->LDS DMA, overlapped with WMMA.
// Epilogue: Q,K stored [b,h,n,d] (Q pre-scaled, log2e folded); V stored
// transposed [b,h,d,n] so the attention PV B-fragments are key-contiguous.

__global__ __launch_bounds__(256) void k_qkv_gemm(
    const bf16* __restrict__ X, const bf16* __restrict__ Wt,
    bf16* __restrict__ Q, bf16* __restrict__ Kb, bf16* __restrict__ Vt) {
    __shared__ __align__(16) bf16 As[2][64 * APAD];
    __shared__ __align__(16) bf16 Bs[2][128 * APAD];
    const int row0 = blockIdx.y * 64;
    const int col0 = blockIdx.x * 128;
    const int tid  = threadIdx.x;
    const int wave = tid >> 5, lane = tid & 31;
    const int wm = wave >> 2, wn = wave & 3;
    const int ar = tid >> 2, ac = (tid & 3) * 8;    // A: 8 bf16 / thread
    const int bn = tid >> 1, bc = (tid & 1) * 16;   // B: 16 bf16 / thread

    auto stage = [&](int b, int kk) {
        unsigned long long ga =
            (unsigned long long)(X + (size_t)(row0 + ar) * DIM + kk + ac);
        ASYNC_LDS_B128_X1(lds_off(&As[b][ar * APAD + ac]), ga);
        unsigned long long gb =
            (unsigned long long)(Wt + (size_t)(col0 + bn) * DIM + kk + bc);
        ASYNC_LDS_B128_X2(lds_off(&Bs[b][bn * APAD + bc]), gb);
    };

    v8f acc[2][2] = {};
    stage(0, 0);
    WAIT_ASYNC0();
    __syncthreads();
    int buf = 0;
    for (int kk = 0; kk < DIM; kk += 32) {
        if (kk + 32 < DIM) stage(buf ^ 1, kk + 32);
        v16bf a0 = load_A(&As[buf][(wm * 32) * APAD], APAD);
        v16bf a1 = load_A(&As[buf][(wm * 32 + 16) * APAD], APAD);
        v16bf b0 = load_BT(&Bs[buf][(wn * 32) * APAD], APAD);
        v16bf b1 = load_BT(&Bs[buf][(wn * 32 + 16) * APAD], APAD);
        acc[0][0] = wmma_bf16(a0, b0, acc[0][0]);
        acc[0][1] = wmma_bf16(a0, b1, acc[0][1]);
        acc[1][0] = wmma_bf16(a1, b0, acc[1][0]);
        acc[1][1] = wmma_bf16(a1, b1, acc[1][1]);
        WAIT_ASYNC0();
        __syncthreads();
        buf ^= 1;
    }
    const int lane15 = lane & 15, laneh = (lane >> 4) << 3;
#pragma unroll
    for (int mt = 0; mt < 2; ++mt) {
#pragma unroll
        for (int nt = 0; nt < 2; ++nt) {
            int col  = col0 + wn * 32 + nt * 16 + lane15;
            int sel  = col / DIM;       // 0=q 1=k 2=v
            int cc   = col % DIM;
            int head = cc / HD, d = cc % HD;
            int mbase = row0 + wm * 32 + mt * 16 + laneh;
            float sc = (sel == 0) ? SCALE_Q : 1.0f;
#pragma unroll
            for (int r = 0; r < 8; ++r) {
                int mrow = mbase + r;
                int b = mrow >> 10, n = mrow & 1023;
                float v = acc[mt][nt][r] * sc;
                size_t hb = (size_t)(b * HEADS + head);
                if (sel == 0)      Q [(hb * NTOK + n) * HD + d] = (bf16)v;
                else if (sel == 1) Kb[(hb * NTOK + n) * HD + d] = (bf16)v;
                else               Vt[(hb * HD + d) * NTOK + n] = (bf16)v;
            }
        }
    }
}

// ---------------- Kernel 3: flash attention ----------------
// grid = 128 (b,head) * 8 qgroups; 8 waves/block, one 16-row query tile/wave.
// K chunk [key][d] (d-contiguous), V chunk pre-transposed [d][key]
// (key-contiguous) -> all fragment loads are b128. Chunks double-buffered and
// streamed by async global->LDS DMA. Softmax in exp2 domain.

__global__ __launch_bounds__(256) void k_attn(
    const bf16* __restrict__ Q, const bf16* __restrict__ Kg,
    const bf16* __restrict__ Vtg, bf16* __restrict__ Oout) {
    __shared__ __align__(16) bf16 Ks[2][KCHUNK * APAD];   // 2 x 20 KB
    __shared__ __align__(16) bf16 Vs[2][HD * VLD];        // 2 x 16.5 KB
    __shared__ __align__(16) bf16 Ps[8][16 * APAD];       // 10 KB
    const int tid  = threadIdx.x;
    const int wave = tid >> 5;
    const int lane = tid & 31;
    const int head_lin = blockIdx.x >> 3;              // b*16+h
    const int qgroup   = blockIdx.x & 7;
    const int qrow0    = (qgroup * 8 + wave) * 16;
    const bf16* Qh  = Q   + (size_t)head_lin * NTOK * HD;
    const bf16* Kh  = Kg  + (size_t)head_lin * NTOK * HD;
    const bf16* Vth = Vtg + (size_t)head_lin * HD * NTOK;

    auto stageKV = [&](int b, int c0) {
        // K: thread t copies key row (c0+t), 64 B
        unsigned long long gk = (unsigned long long)(Kh + (size_t)(c0 + tid) * HD);
        ASYNC_LDS_B128_X4(lds_off(&Ks[b][tid * APAD]), gk);
        // V^T: thread (d = tid>>3, seg = tid&7) copies 32 keys, 64 B
        int d = tid >> 3, seg = (tid & 7) * 32;
        unsigned long long gv =
            (unsigned long long)(Vth + (size_t)d * NTOK + c0 + seg);
        ASYNC_LDS_B128_X4(lds_off(&Vs[b][d * VLD + seg]), gv);
    };

    v16bf qa = load_A(Qh + (size_t)qrow0 * HD, HD);    // scale*log2e folded in
    v8f o0 = {}, o1 = {};
    float mprev[8], lsum[8];
#pragma unroll
    for (int r = 0; r < 8; ++r) { mprev[r] = -1e30f; lsum[r] = 0.f; }

    stageKV(0, 0);
    WAIT_ASYNC0();
    __syncthreads();
    int buf = 0;
    for (int c0 = 0; c0 < NTOK; c0 += KCHUNK) {
        if (c0 + KCHUNK < NTOK) stageKV(buf ^ 1, c0 + KCHUNK);

        for (int j = 0; j < KCHUNK; j += 32) {
            v16bf bk0 = load_BT(&Ks[buf][j * APAD], APAD);
            v16bf bk1 = load_BT(&Ks[buf][(j + 16) * APAD], APAD);
            v8f s0 = {}, s1 = {};
            s0 = wmma_bf16(qa, bk0, s0);
            s1 = wmma_bf16(qa, bk1, s1);

            bf16* ps = &Ps[wave][0];
#pragma unroll
            for (int r = 0; r < 8; ++r) {
                float tmax = fmaxf(s0[r], s1[r]);
#pragma unroll
                for (int off = 1; off < 16; off <<= 1)
                    tmax = fmaxf(tmax, __shfl_xor(tmax, off, 32));
                float mnew = fmaxf(mprev[r], tmax);
                float corr = __builtin_amdgcn_exp2f(mprev[r] - mnew);
                float p0 = __builtin_amdgcn_exp2f(s0[r] - mnew);
                float p1 = __builtin_amdgcn_exp2f(s1[r] - mnew);
                float rsum = p0 + p1;
#pragma unroll
                for (int off = 1; off < 16; off <<= 1)
                    rsum += __shfl_xor(rsum, off, 32);
                lsum[r] = lsum[r] * corr + rsum;
                mprev[r] = mnew;
                o0[r] *= corr;
                o1[r] *= corr;
                int prow = r + ((lane >> 4) << 3);
                ps[prow * APAD + (lane & 15)]      = (bf16)p0;
                ps[prow * APAD + 16 + (lane & 15)] = (bf16)p1;
            }
            __builtin_amdgcn_wave_barrier();   // wave-synchronous LDS bounce
            v16bf pa  = load_A(ps, APAD);
            v16bf vb0 = load_BT(&Vs[buf][j], VLD);              // d cols 0..15
            v16bf vb1 = load_BT(&Vs[buf][16 * VLD + j], VLD);   // d cols 16..31
            o0 = wmma_bf16(pa, vb0, o0);
            o1 = wmma_bf16(pa, vb1, o1);
            __builtin_amdgcn_wave_barrier();
        }
        WAIT_ASYNC0();
        __syncthreads();
        buf ^= 1;
    }
    // normalize + scatter to attn buffer [8192][512] bf16
    const int b = head_lin >> 4, h = head_lin & 15;
    const int colb = h * HD + (lane & 15);
#pragma unroll
    for (int r = 0; r < 8; ++r) {
        int mrow = qrow0 + r + ((lane >> 4) << 3);
        float inv = 1.0f / lsum[r];
        size_t grow = (size_t)(b * NTOK + mrow) * DIM;
        Oout[grow + colb]      = (bf16)(o0[r] * inv);
        Oout[grow + colb + 16] = (bf16)(o1[r] * inv);
    }
}

// ---------------- Kernel 4: projection GEMM + bias -> fp32 ----------------
// Same 64x128 tiling + async double buffering; B from pre-transposed Wpt.

__global__ __launch_bounds__(256) void k_proj_gemm(
    const bf16* __restrict__ A, const bf16* __restrict__ Wt,
    const float* __restrict__ bias, float* __restrict__ Out) {
    __shared__ __align__(16) bf16 As[2][64 * APAD];
    __shared__ __align__(16) bf16 Bs[2][128 * APAD];
    const int row0 = blockIdx.y * 64;
    const int col0 = blockIdx.x * 128;
    const int tid  = threadIdx.x;
    const int wave = tid >> 5, lane = tid & 31;
    const int wm = wave >> 2, wn = wave & 3;
    const int ar = tid >> 2, ac = (tid & 3) * 8;
    const int bn = tid >> 1, bc = (tid & 1) * 16;

    auto stage = [&](int b, int kk) {
        unsigned long long ga =
            (unsigned long long)(A + (size_t)(row0 + ar) * DIM + kk + ac);
        ASYNC_LDS_B128_X1(lds_off(&As[b][ar * APAD + ac]), ga);
        unsigned long long gb =
            (unsigned long long)(Wt + (size_t)(col0 + bn) * DIM + kk + bc);
        ASYNC_LDS_B128_X2(lds_off(&Bs[b][bn * APAD + bc]), gb);
    };

    v8f acc[2][2] = {};
    stage(0, 0);
    WAIT_ASYNC0();
    __syncthreads();
    int buf = 0;
    for (int kk = 0; kk < DIM; kk += 32) {
        if (kk + 32 < DIM) stage(buf ^ 1, kk + 32);
        v16bf a0 = load_A(&As[buf][(wm * 32) * APAD], APAD);
        v16bf a1 = load_A(&As[buf][(wm * 32 + 16) * APAD], APAD);
        v16bf b0 = load_BT(&Bs[buf][(wn * 32) * APAD], APAD);
        v16bf b1 = load_BT(&Bs[buf][(wn * 32 + 16) * APAD], APAD);
        acc[0][0] = wmma_bf16(a0, b0, acc[0][0]);
        acc[0][1] = wmma_bf16(a0, b1, acc[0][1]);
        acc[1][0] = wmma_bf16(a1, b0, acc[1][0]);
        acc[1][1] = wmma_bf16(a1, b1, acc[1][1]);
        WAIT_ASYNC0();
        __syncthreads();
        buf ^= 1;
    }
    const int lane15 = lane & 15, laneh = (lane >> 4) << 3;
#pragma unroll
    for (int mt = 0; mt < 2; ++mt) {
#pragma unroll
        for (int nt = 0; nt < 2; ++nt) {
            int col   = col0 + wn * 32 + nt * 16 + lane15;
            int mbase = row0 + wm * 32 + mt * 16 + laneh;
            float bv = bias[col];
#pragma unroll
            for (int r = 0; r < 8; ++r)
                Out[(size_t)(mbase + r) * DIM + col] = acc[mt][nt][r] + bv;
        }
    }
}

// ---------------- host-side launcher ----------------

extern "C" void kernel_launch(void* const* d_in, const int* in_sizes, int n_in,
                              void* d_out, int out_size, void* d_ws, size_t ws_size,
                              hipStream_t stream) {
    (void)in_sizes; (void)n_in; (void)out_size; (void)ws_size;
    const float* x      = (const float*)d_in[0];
    const float* w_qkv  = (const float*)d_in[1];
    const float* w_proj = (const float*)d_in[2];
    const float* b_proj = (const float*)d_in[3];
    float* out = (float*)d_out;

    char* ws = (char*)d_ws;
    const size_t nX  = (size_t)M_TOTAL * DIM;      // 4,194,304
    const size_t nWq = (size_t)DIM * 3 * DIM;      //   786,432
    const size_t nWp = (size_t)DIM * DIM;          //   262,144
    bf16* xb     = (bf16*)ws;  ws += nX  * 2;
    bf16* wqkvt  = (bf16*)ws;  ws += nWq * 2;      // [1536][512] transposed
    bf16* wprojt = (bf16*)ws;  ws += nWp * 2;      // [512][512]  transposed
    bf16* Qb     = (bf16*)ws;  ws += nX * 2;       // [B,h,n,d]
    bf16* Kb     = (bf16*)ws;  ws += nX * 2;       // [B,h,n,d]
    bf16* Vtb    = (bf16*)ws;  ws += nX * 2;       // [B,h,d,n] (transposed)
    bf16* attnb  = (bf16*)ws;  ws += nX * 2;       // [8192,512]

    k_cvt_bf16<<<2048, 256, 0, stream>>>(x, xb, (int)nX);
    k_cvt_t<<<1024, 256, 0, stream>>>(w_qkv, wqkvt, (int)nWq);
    k_cvt_t<<<512, 256, 0, stream>>>(w_proj, wprojt, (int)nWp);

    dim3 gQKV(3 * DIM / 128, M_TOTAL / 64);   // 12 x 128
    k_qkv_gemm<<<gQKV, 256, 0, stream>>>(xb, wqkvt, Qb, Kb, Vtb);

    k_attn<<<BATCH * HEADS * 8, 256, 0, stream>>>(Qb, Kb, Vtb, attnb);

    dim3 gP(DIM / 128, M_TOTAL / 64);         // 4 x 128
    k_proj_gemm<<<gP, 256, 0, stream>>>(attnb, wprojt, b_proj, out);
}
